// SecondOrderInteraction_82901458747778
// MI455X (gfx1250) — matrix operand order repaired and checked by reading, not compiled, not run on hardware
//
#include <hip/hip_runtime.h>
#include <stdint.h>

// ---------------------------------------------------------------------------
// FM second-order interaction, MI455X (gfx1250), wave32 + WMMA bf16.
//
// out[b] = 0.5 * ( sum_f (x@v)[b,f]^2  -  sum_n x[b,n]^2 * w[n] ),
//   w[n] = sum_f v[n,f]^2.
//
// One bf16 GEMM (fp32 accumulate, v_wmma_f32_16x16x32_bf16) + fused x^2.w.
// v pre-swizzled once into bf16 B-fragments (2 MB, L2 resident).
// B tile: global->LDS via CDNA5 async copy (global_load_async_to_lds_b128,
// ASYNCcnt) -- zero VGPR staging. x tile: register-pipelined (needed in VGPRs
// for the fused x^2.w term and bf16 conversion), overlapping the WMMA burst.
// ---------------------------------------------------------------------------

typedef __attribute__((ext_vector_type(16))) __bf16 v16bf;
typedef __attribute__((ext_vector_type(8)))  float  v8f;
typedef __attribute__((address_space(3))) unsigned char lds_char;

#define B_ROWS   16384
#define N_DIM    4096
#define F_DIM    256
#define M_TILE   64
#define K_TILE   64
#define NUM_ITERS (N_DIM / K_TILE)   // 64

// LDS geometry
#define XS_PITCH_B  144              // 64 bf16 + 8 pad elems (16B-aligned, conflict-free)
#define XS_BYTES    (M_TILE * XS_PITCH_B)              // 9216
#define BS_SLOT_U32 12               // 8 payload u32 + 4 pad -> 48 B slot (16-aligned)
#define BS_BYTES    (2 * 16 * 32 * BS_SLOT_U32 * 4)    // 49152

union FragAB { uint4 q[2]; v16bf v; };

__device__ __forceinline__ uint32_t pk_bf16(float a, float b) {
  uint32_t ua = __builtin_bit_cast(uint32_t, a);
  uint32_t ub = __builtin_bit_cast(uint32_t, b);
  ua = (ua + 0x7FFFu + ((ua >> 16) & 1u)) >> 16;   // round-to-nearest-even
  ub = (ub + 0x7FFFu + ((ub >> 16) & 1u)) >> 16;
  return ua | (ub << 16);
}

__device__ __forceinline__ uint32_t lds_offset_of(const void* p) {
  return (uint32_t)(uintptr_t)(lds_char*)p;     // generic -> LDS addrspace cast
}

// Async copy 32 B global -> LDS (two b128, offset applies to both addresses).
__device__ __forceinline__ void async_copy_32B(uint32_t lds_addr, uint64_t gaddr) {
  asm volatile("global_load_async_to_lds_b128 %0, %1, off\n\t"
               "global_load_async_to_lds_b128 %0, %1, off offset:16"
               :: "v"(lds_addr), "v"(gaddr) : "memory");
}

__device__ __forceinline__ void wait_asynccnt0() {
  asm volatile("s_wait_asynccnt 0x0" ::: "memory");
}

// ---------------------------------------------------------------------------
// Prep 1: w[n] = sum_f v[n,f]^2.  One block per n, coalesced.
// ---------------------------------------------------------------------------
__global__ void fm_prep_w(const float* __restrict__ v, float* __restrict__ w) {
  const int n = blockIdx.x;
  const float val = v[(size_t)n * F_DIM + threadIdx.x];
  float s = val * val;
#pragma unroll
  for (int m = 16; m >= 1; m >>= 1) s += __shfl_xor(s, m, 32);
  __shared__ float red[8];
  if ((threadIdx.x & 31) == 0) red[threadIdx.x >> 5] = s;
  __syncthreads();
  if (threadIdx.x == 0) {
    float t = 0.f;
#pragma unroll
    for (int i = 0; i < 8; ++i) t += red[i];
    w[n] = t;
  }
}

// ---------------------------------------------------------------------------
// Prep 2: swizzle v (fp32 [n][f] row-major) into bf16 WMMA-B fragments:
//   bswz[kb][ftg][lane][j] : u32 = {v[k][f], v[k+1][f]} bf16-packed, where
//   k = kb*32 + (lane<16 ? 0 : 16) + 2j,  f = ftg*16 + (lane&15).
// ---------------------------------------------------------------------------
__global__ void fm_prep_bswz(const float* __restrict__ v, uint32_t* __restrict__ bswz) {
  const int g    = blockIdx.x * blockDim.x + threadIdx.x;   // [0, 128*16*32)
  const int lane = g & 31;
  const int ftg  = (g >> 5) & 15;
  const int kb   = g >> 9;
  const int f     = ftg * 16 + (lane & 15);
  const int kbase = kb * 32 + ((lane >= 16) ? 16 : 0);
  uint32_t o[8];
#pragma unroll
  for (int j = 0; j < 8; ++j) {
    const float a = v[(size_t)(kbase + 2 * j)     * F_DIM + f];
    const float b = v[(size_t)(kbase + 2 * j + 1) * F_DIM + f];
    o[j] = pk_bf16(a, b);
  }
  uint32_t* dst = bswz + (size_t)g * 8;
  *(uint4*)(dst)     = make_uint4(o[0], o[1], o[2], o[3]);
  *(uint4*)(dst + 4) = make_uint4(o[4], o[5], o[6], o[7]);
}

// ---------------------------------------------------------------------------
// Main: 256 threads = 8 waves. Wave w -> row group rg=w>>1 (16 rows), f-chunk
// fc=w&1 (128 cols -> 8 N-tiles). Acc = 8 x v8f = 64 VGPRs/wave.
// ---------------------------------------------------------------------------
__global__ void __launch_bounds__(256) fm_main(
    const float* __restrict__ x, const uint32_t* __restrict__ bswz,
    const float* __restrict__ w, float* __restrict__ out)
{
  __shared__ __align__(16) unsigned char smem[BS_BYTES + XS_BYTES + 512];
  uint32_t* Bs      = (uint32_t*)smem;
  unsigned char* Xs = smem + BS_BYTES;
  float* t2_lds = (float*)(smem + BS_BYTES + XS_BYTES);
  float* sq_lds = t2_lds + 64;

  const int tid  = threadIdx.x;
  const int lane = tid & 31;
  const int wave = tid >> 5;
  const int rg   = wave >> 1;              // 0..3
  const int fc   = wave & 1;               // 0..1
  const int wgRow0 = blockIdx.x * M_TILE;

  if (tid < 64) { t2_lds[tid] = 0.f; sq_lds[tid] = 0.f; }

  // x loader mapping: thread -> rows {lrow + 16i}, float4 column lcol4
  const int lrow  = tid >> 4;              // 0..15
  const int lcol4 = tid & 15;              // 0..15
  const float* xbase = x + (size_t)(wgRow0 + lrow) * N_DIM + lcol4 * 4;
  float t2acc[4] = {0.f, 0.f, 0.f, 0.f};

  v8f acc[8] = {};                         // fp32 C fragments

  // A-fragment lane addressing (contiguous k: lane half h holds k 16h..16h+15, same as bswz)
  const int arow_off = (rg * 16 + (lane & 15)) * XS_PITCH_B;
  const int akoff    = (lane >= 16) ? 32 : 0;   // bytes

  // B async-copy: thread owns 4 (ks,ftg,lane) triples = tid + 256*p
  uint32_t blds[4];      // LDS byte address of slot
  uint64_t bsrc[4];      // global byte address (tile 0)
#pragma unroll
  for (int p = 0; p < 4; ++p) {
    const int trip = tid + 256 * p;
    const int bl  = trip & 31;
    const int bf  = (trip >> 5) & 15;
    const int bks = trip >> 9;
    const int slot = (bks * 16 + bf) * 32 + bl;
    blds[p] = lds_offset_of(Bs) + (uint32_t)slot * (BS_SLOT_U32 * 4);
    bsrc[p] = (uint64_t)(uintptr_t)(bswz + (size_t)slot * 8);
  }

  // -------- x register pipeline (tile it+1 in flight during compute) --------
  float4 xr[4];
  float4 wr;
  wr = *((const float4*)(w) + lcol4);
#pragma unroll
  for (int i = 0; i < 4; ++i) xr[i] = *(const float4*)(xbase + (size_t)i * 16 * N_DIM);

  for (int it = 0; it < NUM_ITERS; ++it) {
    __syncthreads();                       // previous tile fully consumed

    // ---- B tile: async global->LDS, zero VGPR staging (ASYNCcnt) ----
    const uint64_t bofs = (uint64_t)it * (2 * 16 * 32 * 8 * 4);  // bytes per K-tile
#pragma unroll
    for (int p = 0; p < 4; ++p) async_copy_32B(blds[p], bsrc[p] + bofs);

    // ---- x tile: store staged regs to LDS, fused x^2.w ----
#pragma unroll
    for (int i = 0; i < 4; ++i) {
      const float4 xq = xr[i];
      t2acc[i] += xq.x * xq.x * wr.x + xq.y * xq.y * wr.y
                + xq.z * xq.z * wr.z + xq.w * xq.w * wr.w;
      const uint32_t p0 = pk_bf16(xq.x, xq.y);
      const uint32_t p1 = pk_bf16(xq.z, xq.w);
      *(uint2*)(Xs + (lrow + i * 16) * XS_PITCH_B + lcol4 * 8) = make_uint2(p0, p1);
    }

    wait_asynccnt0();                      // this wave's B copies landed in LDS
    __syncthreads();                       // all waves: Xs + Bs ready

    // ---- issue global loads for x tile it+1 (overlap with WMMA burst) ----
    if (it + 1 < NUM_ITERS) {
      const int k1 = (it + 1) * K_TILE;
      wr = *((const float4*)(w + k1) + lcol4);
#pragma unroll
      for (int i = 0; i < 4; ++i) {
        const float* src = xbase + (size_t)i * 16 * N_DIM + k1;
        xr[i] = *(const float4*)src;
        if (it + 2 < NUM_ITERS) __builtin_prefetch(src + K_TILE, 0, 1);  // global_prefetch_b8
      }
    }

    // ---- compute: per k-step preload fa + 8 fb, then 8 WMMAs back-to-back ----
#pragma unroll
    for (int ks = 0; ks < 2; ++ks) {
      FragAB fa;
      const unsigned char* ap = Xs + arow_off + ks * 64 + akoff;
      fa.q[0] = *(const uint4*)(ap);        // K 0..7   (hi lanes: 16..23)
      fa.q[1] = *(const uint4*)(ap + 16);   // K 8..15  (hi lanes: 24..31)
      FragAB fb[8];
#pragma unroll
      for (int ft = 0; ft < 8; ++ft) {
        const uint32_t* bp = Bs + ((ks * 16 + fc * 8 + ft) * 32 + lane) * BS_SLOT_U32;
        fb[ft].q[0] = *(const uint4*)(bp);
        fb[ft].q[1] = *(const uint4*)(bp + 4);
      }
#pragma unroll
      for (int ft = 0; ft < 8; ++ft)
        acc[ft] = __builtin_amdgcn_wmma_f32_16x16x32_bf16(
            false, fa.v, false, fb[ft].v, (short)0, acc[ft], false, false);
    }
  }

  // ---- epilogue: sum_f xv^2 per row ----
  // C layout: lanes 0-15 hold rows rg*16+reg (N=lane), lanes 16-31 rows rg*16+8+reg.
  float s[8];
#pragma unroll
  for (int r = 0; r < 8; ++r) {
    float t = 0.f;
#pragma unroll
    for (int ft = 0; ft < 8; ++ft) { const float e = acc[ft][r]; t += e * e; }
    s[r] = t;
  }
#pragma unroll
  for (int r = 0; r < 8; ++r) {
#pragma unroll
    for (int m = 8; m >= 1; m >>= 1) s[r] += __shfl_xor(s[r], m, 32);  // 16-lane halves
  }

  if (lane == 0 || lane == 16) {
    const int rbase = rg * 16 + ((lane == 16) ? 8 : 0);
#pragma unroll
    for (int r = 0; r < 8; ++r) atomicAdd(&sq_lds[rbase + r], s[r]);   // ds_add_f32
  }
#pragma unroll
  for (int i = 0; i < 4; ++i) atomicAdd(&t2_lds[lrow + i * 16], t2acc[i]);
  __syncthreads();

  if (tid < 64) out[wgRow0 + tid] = 0.5f * (sq_lds[tid] - t2_lds[tid]);
}

// ---------------------------------------------------------------------------
// Launch. Workspace: [0,16KB) w[4096] fp32; [16KB, 16KB+2MB) bswz.
// ---------------------------------------------------------------------------
extern "C" void kernel_launch(void* const* d_in, const int* in_sizes, int n_in,
                              void* d_out, int out_size, void* d_ws, size_t ws_size,
                              hipStream_t stream) {
  (void)in_sizes; (void)n_in; (void)out_size; (void)ws_size;
  const float* x = (const float*)d_in[0];        // [16384, 4096] fp32
  const float* v = (const float*)d_in[1];        // [4096, 256]  fp32
  float*    out  = (float*)d_out;                // [16384] fp32
  float*    w    = (float*)d_ws;
  uint32_t* bswz = (uint32_t*)((char*)d_ws + 16384);

  fm_prep_w   <<<N_DIM,                 F_DIM, 0, stream>>>(v, w);
  fm_prep_bswz<<<(128 * 16 * 32) / 256, 256,   0, stream>>>(v, bswz);
  fm_main     <<<B_ROWS / M_TILE,       256,   0, stream>>>(x, bswz, w, out);
}
